// LightAttention_13039520710699
// MI455X (gfx1250) — compile-verified, matrix operand order checked by reading, not run on hardware
//
#include <hip/hip_runtime.h>
#include <hip/hip_bf16.h>
#include <math.h>

typedef __bf16 bf16_t;
typedef __attribute__((ext_vector_type(16))) __bf16 v16bf;
typedef __attribute__((ext_vector_type(4)))  __bf16 v4bf;
typedef __attribute__((ext_vector_type(8)))  float  v8f;
typedef __attribute__((ext_vector_type(4)))  float  f32x4;

#define BATCH 4
#define TLEN  4096
#define DDIM  512
#define NKCHUNK 128

union FragBF { v16bf v; f32x4 f4[2]; };

// Load a 16x32 bf16 A/B WMMA fragment: per-lane two contiguous runs of 8 bf16
// at element offsets kb and kb+16 (kb = 8*(lane>>4)).  16-byte aligned.
__device__ __forceinline__ v16bf load_frag_bf16(const bf16_t* p, int kb) {
  FragBF f;
  f.f4[0] = *(const f32x4*)(p + kb);
  f.f4[1] = *(const f32x4*)(p + kb + 16);
  return f.v;
}

// ---- Kernel A: x f32 [B,T,D] -> xbf (row-major bf16) + xT (bf16 [B,D,T]) ---
// LDS-tiled 32x32 transpose; all global accesses coalesced.
__global__ __launch_bounds__(256) void convert_transpose_kernel(
    const float* __restrict__ x, bf16_t* __restrict__ xbf, bf16_t* __restrict__ xT) {
  const int tile = blockIdx.x;              // B * (TLEN/32) * (DDIM/32) = 8192
  const int dt = tile & 15;                 // DDIM/32 = 16
  const int tt = (tile >> 4) & 127;         // TLEN/32 = 128
  const int b  = tile >> 11;
  const int d0 = dt * 32, t0 = tt * 32;

  __shared__ bf16_t tileS[32][34];          // +2 pad vs bank conflicts

  const int r  = threadIdx.x >> 3;          // 0..31: t row within tile
  const int c4 = (threadIdx.x & 7) * 4;     // d col group of 4
  f32x4 v = *(const f32x4*)(x + ((size_t)b * TLEN + t0 + r) * DDIM + d0 + c4);
  v4bf bv;
#pragma unroll
  for (int j = 0; j < 4; ++j) { bv[j] = (bf16_t)v[j]; tileS[r][c4 + j] = bv[j]; }
  *(v4bf*)(xbf + ((size_t)b * TLEN + t0 + r) * DDIM + d0 + c4) = bv;
  __syncthreads();

  const int dr = threadIdx.x >> 3;          // 0..31: d row of xT
  const int tc = (threadIdx.x & 7) * 4;     // t group of 4
  v4bf ov;
#pragma unroll
  for (int j = 0; j < 4; ++j) ov[j] = tileS[tc + j][dr];
  *(v4bf*)(xT + ((size_t)b * DDIM + d0 + dr) * TLEN + t0 + tc) = ov;
}

// ---- Kernel B: generic f32 -> bf16 convert (for Wq, Wk) --------------------
__global__ __launch_bounds__(256) void cvt_bf16_kernel(
    const float* __restrict__ in, bf16_t* __restrict__ out, int n) {
  int i = blockIdx.x * 256 + threadIdx.x;
  if (i < n) out[i] = (bf16_t)in[i];
}

// ---- Kernel 1: Q = x Wq^T + bq ; K = x Wk^T + bk (all-bf16 operands) -------
__global__ __launch_bounds__(128) void qk_proj_kernel(
    const bf16_t* __restrict__ xbf,
    const bf16_t* __restrict__ Wqb, const float* __restrict__ bq,
    const bf16_t* __restrict__ Wkb, const float* __restrict__ bk,
    bf16_t* __restrict__ Qb, bf16_t* __restrict__ Kb) {
  const int lane = threadIdx.x & 31;
  const int wave = threadIdx.x >> 5;
  const int task = blockIdx.x * 4 + wave;      // 32768 tasks
  const int ntile = task & 31;                 // e tile (32)
  const int mtile = task >> 5;                 // row tile (1024)
  const int row0 = mtile * 16;
  const int e0   = ntile * 16;
  const int p = lane & 15;
  const int h = lane >> 4;
  const int kb = 8 * h;

  const bf16_t* xrow  = xbf + (size_t)(row0 + p) * DDIM;
  const bf16_t* wqrow = Wqb + (size_t)(e0 + p) * DDIM;
  const bf16_t* wkrow = Wkb + (size_t)(e0 + p) * DDIM;

  v8f cq = {}; v8f ck = {};
#pragma unroll
  for (int dd = 0; dd < DDIM; dd += 32) {
    v16bf a  = load_frag_bf16(xrow + dd, kb);
    v16bf fq = load_frag_bf16(wqrow + dd, kb);
    v16bf fk = load_frag_bf16(wkrow + dd, kb);
    cq = __builtin_amdgcn_wmma_f32_16x16x32_bf16(false, a, false, fq, (short)0, cq, false, false);
    ck = __builtin_amdgcn_wmma_f32_16x16x32_bf16(false, a, false, fk, (short)0, ck, false, false);
  }
  const float biasq = bq[e0 + p];
  const float biask = bk[e0 + p];
#pragma unroll
  for (int r = 0; r < 8; ++r) {
    int m = row0 + r + 8 * h;                  // C/D layout: row = r + 8h, col = p
    Qb[(size_t)m * DDIM + e0 + p] = (bf16_t)(cq[r] + biasq);
    Kb[(size_t)m * DDIM + e0 + p] = (bf16_t)(ck[r] + biask);
  }
}

// ---- Kernel 2: fused flash attention + residual ----------------------------
__global__ __launch_bounds__(256) void flash_kernel(
    const float* __restrict__ x, const float* __restrict__ gamma,
    const bf16_t* __restrict__ Qb, const bf16_t* __restrict__ Kb,
    const bf16_t* __restrict__ xT, float* __restrict__ out) {
  const int tid  = threadIdx.x;
  const int lane = tid & 31;
  const int wave = tid >> 5;                   // 0..7
  const int b  = blockIdx.x >> 8;              // 256 query tiles per batch
  const int t0 = (blockIdx.x & 255) * 16;
  const int p = lane & 15;
  const int h = lane >> 4;
  const int kb = 8 * h;
  const size_t rowBase = (size_t)b * TLEN;

  __shared__ float  Sbuf[16][NKCHUNK];
  __shared__ __attribute__((aligned(16))) bf16_t Pbuf[16][NKCHUNK];
  __shared__ float  red[16][16];
  __shared__ float  mstate[16], lstate[16], alpha_s[16], mnew_s[16];

  if (tid < 16) { mstate[tid] = -1e30f; lstate[tid] = 0.f; }

  // Preload this query tile's Q as 16 A-fragments (held in registers).
  const bf16_t* qrow = Qb + (rowBase + t0 + p) * DDIM;
  v16bf Qfrag[16];
#pragma unroll
  for (int i = 0; i < 16; ++i) Qfrag[i] = load_frag_bf16(qrow + i * 32, kb);

  v8f acc[4] = {};                             // e slice [wave*64, wave*64+64)
  const int e0 = wave * 64;
  const float scale = 0.044194173824159216f;   // 1/sqrt(512)
  const bf16_t* vbase = xT + ((size_t)b * DDIM + e0 + p) * TLEN;

  __syncthreads();

  for (int key0 = 0; key0 < TLEN; key0 += NKCHUNK) {
    // ---- scores: wave w computes 16x16 tile for keys key0 + w*16 ----
    const bf16_t* krow = Kb + (rowBase + key0 + wave * 16 + p) * DDIM;
    v8f s = {};
#pragma unroll
    for (int dd = 0; dd < DDIM; dd += 32) {
      v16bf kf = load_frag_bf16(krow + dd, kb);
      s = __builtin_amdgcn_wmma_f32_16x16x32_bf16(false, Qfrag[dd >> 5], false, kf, (short)0, s, false, false);
    }
#pragma unroll
    for (int r = 0; r < 8; ++r) Sbuf[r + 8 * h][wave * 16 + p] = s[r] * scale;
    __syncthreads();

    // ---- online softmax over the 16x128 chunk: 16 threads/row, 8 cols each
    const int r  = tid >> 4;
    const int cg = tid & 15;
    float lmax = -1e30f;
#pragma unroll
    for (int j = 0; j < 8; ++j) lmax = fmaxf(lmax, Sbuf[r][cg * 8 + j]);
    red[r][cg] = lmax;
    __syncthreads();
    if (cg == 0) {
      float cm = red[r][0];
#pragma unroll
      for (int j = 1; j < 16; ++j) cm = fmaxf(cm, red[r][j]);
      float mold = mstate[r];
      float mnew = fmaxf(mold, cm);
      mnew_s[r]  = mnew;
      alpha_s[r] = __expf(mold - mnew);
      mstate[r]  = mnew;
    }
    __syncthreads();
    const float mnew = mnew_s[r];
    float psum = 0.f;
#pragma unroll
    for (int j = 0; j < 8; ++j) {
      float pv = __expf(Sbuf[r][cg * 8 + j] - mnew);
      Pbuf[r][cg * 8 + j] = (bf16_t)pv;
      psum += pv;
    }
    red[r][cg] = psum;
    __syncthreads();
    if (cg == 0) {
      float rs = 0.f;
#pragma unroll
      for (int j = 0; j < 16; ++j) rs += red[r][j];
      lstate[r] = lstate[r] * alpha_s[r] + rs;
    }
    __syncthreads();

    // ---- PV: rescale accumulators, then acc += P @ V over the 128 keys ----
#pragma unroll
    for (int e = 0; e < 4; ++e)
#pragma unroll
      for (int rr = 0; rr < 8; ++rr) acc[e][rr] *= alpha_s[rr + 8 * h];

#pragma unroll
    for (int kk = 0; kk < 4; ++kk) {
      v16bf pf = load_frag_bf16(&Pbuf[p][0], kk * 32 + kb);   // A: row = p
#pragma unroll
      for (int e = 0; e < 4; ++e) {
        const bf16_t* vcol = vbase + (size_t)(e * 16) * TLEN + key0 + kk * 32;
        v16bf vf = load_frag_bf16(vcol, kb);                  // B: col = p
        acc[e] = __builtin_amdgcn_wmma_f32_16x16x32_bf16(false, pf, false, vf, (short)0, acc[e], false, false);
      }
    }
    __syncthreads();   // guards Sbuf/Pbuf/alpha reuse in next iteration
  }

  // ---- epilogue: out = x + gamma * acc / l ----
  const float g = gamma[0];
#pragma unroll
  for (int e = 0; e < 4; ++e) {
#pragma unroll
    for (int rr = 0; rr < 8; ++rr) {
      int m = rr + 8 * h;
      size_t idx = (rowBase + t0 + m) * DDIM + e0 + e * 16 + p;
      out[idx] = x[idx] + g * (acc[e][rr] / lstate[m]);
    }
  }
}

extern "C" void kernel_launch(void* const* d_in, const int* in_sizes, int n_in,
                              void* d_out, int out_size, void* d_ws, size_t ws_size,
                              hipStream_t stream) {
  const float* x     = (const float*)d_in[0];
  const float* Wq    = (const float*)d_in[1];
  const float* bq    = (const float*)d_in[2];
  const float* Wk    = (const float*)d_in[3];
  const float* bk    = (const float*)d_in[4];
  const float* gamma = (const float*)d_in[5];
  float* out = (float*)d_out;

  const size_t NROWS = (size_t)BATCH * TLEN;      // 16384
  bf16_t* Qb  = (bf16_t*)d_ws;                    // 16 MB
  bf16_t* Kb  = Qb + NROWS * DDIM;                // 16 MB
  bf16_t* xT  = Kb + NROWS * DDIM;                // 16 MB
  bf16_t* xbf = xT + NROWS * DDIM;                // 16 MB
  bf16_t* Wqb = xbf + NROWS * DDIM;               // 0.5 MB
  bf16_t* Wkb = Wqb + (size_t)DDIM * DDIM;        // 0.5 MB

  // 1) one pass over x: row-major bf16 copy + transposed bf16 copy
  convert_transpose_kernel<<<BATCH * (TLEN / 32) * (DDIM / 32), 256, 0, stream>>>(x, xbf, xT);
  // 2) weights to bf16
  cvt_bf16_kernel<<<(DDIM * DDIM) / 256, 256, 0, stream>>>(Wq, Wqb, DDIM * DDIM);
  cvt_bf16_kernel<<<(DDIM * DDIM) / 256, 256, 0, stream>>>(Wk, Wkb, DDIM * DDIM);
  // 3) Q/K projections (32768 wave tasks, 4 waves per block)
  qk_proj_kernel<<<8192, 128, 0, stream>>>(xbf, Wqb, bq, Wkb, bk, Qb, Kb);
  // 4) fused attention (one block per 16-query tile)
  flash_kernel<<<BATCH * (TLEN / 16), 256, 0, stream>>>(x, gamma, Qb, Kb, xT, out);
}
